// GNN_29635274342453
// MI455X (gfx1250) — matrix operand order repaired and checked by reading, not compiled, not run on hardware
//
#include <hip/hip_runtime.h>

#define NNODES 50000
#define NEDGES 400000
#define NB     128
#define FIN    64
#define EDIMC  8
#define EMBC   64
#define NH     4
#define HCC    256
#define NL     3
#define DNC    256

typedef __attribute__((ext_vector_type(16))) _Float16 v16h;
typedef __attribute__((ext_vector_type(8)))  float    v8f;

// ---- order-preserving float <-> uint key for atomicMax on floats ----
__device__ __forceinline__ unsigned fkey(float f) {
  int i = __float_as_int(f);
  return (i >= 0) ? ((unsigned)i | 0x80000000u) : ~((unsigned)i);
}
__device__ __forceinline__ float finv(unsigned k) {
  return (k & 0x80000000u) ? __int_as_float((int)(k & 0x7fffffffu))
                           : __int_as_float((int)(~k));
}

// CDNA5 async Global->LDS copy (ASYNCcnt-tracked, no VGPR round trip).
// vdst = LDS byte offset (32-bit VGPR), vaddr = 64-bit global address.
__device__ __forceinline__ void async_copy_b128(unsigned lds_off, const float* gaddr) {
  asm volatile("global_load_async_to_lds_b128 %0, %1, off"
               :: "v"(lds_off), "v"(gaddr) : "memory");
}
__device__ __forceinline__ unsigned lds_addr(const void* p) {
  return (unsigned)(size_t)p;   // generic shared ptr: low 32 bits = LDS offset
}

// ============================================================================
// WMMA GEMM with async double-buffered LDS staging:
//   C[M,Nout] = act(A[M,K] @ W[K,Nout] + bias)
// K in {64,256} (template, fully unrolled), Nout multiple of 64,
// M multiple of 16 (tiles always full: 50000 = 3125*16).
// Block = 128 threads (4 waves); tile = 16 rows x 64 cols;
// each wave owns one 16x16 WMMA accumulator.
// Per chunk each wave issues 5 async b128 loads (1 A + 4 W); chunk k+1 is
// prefetched while chunk k computes; s_wait_asynccnt 5 retires chunk k only.
// ============================================================================
template <int K>
__global__ __launch_bounds__(128)
void wmma_gemm_async(const float* __restrict__ A, const float* __restrict__ W,
                     const float* __restrict__ bias, float* __restrict__ C,
                     int Nout, int doRelu)
{
  constexpr int NK = K / 32;
  __shared__ float As[2][16 * 32];
  __shared__ float Bs[2][32 * 64];
  const int tid  = threadIdx.x;
  const int wave = tid >> 5;
  const int lane = tid & 31;
  const int row0 = blockIdx.x * 16;
  const int col0 = blockIdx.y * 64;

  auto issue = [&](int kc, int b) {
    // A tile 16x32: 512 floats = 128 threads x 1 float4, coalesced
    {
      int r = tid >> 3;
      int c = (tid & 7) << 2;
      async_copy_b128(lds_addr(&As[b][tid << 2]),
                      A + (size_t)(row0 + r) * K + kc * 32 + c);
    }
    // W tile 32x64: 2048 floats = 128 threads x 4 float4, coalesced
    #pragma unroll
    for (int i = 0; i < 4; ++i) {
      int idx = ((i << 7) + tid) << 2;
      int r = idx >> 6, c = idx & 63;
      async_copy_b128(lds_addr(&Bs[b][idx]),
                      W + (size_t)(kc * 32 + r) * Nout + col0 + c);
    }
  };

  v8f acc = {};
  issue(0, 0);
  #pragma unroll
  for (int kc = 0; kc < NK; ++kc) {
    const int b = kc & 1;
    if (kc + 1 < NK) {
      issue(kc + 1, b ^ 1);
      asm volatile("s_wait_asynccnt 0x5" ::: "memory");  // retire chunk kc only
    } else {
      asm volatile("s_wait_asynccnt 0x0" ::: "memory");
    }
    __syncthreads();

    // build f16 fragments per CDNA5 ISA register layouts
    v16h a, bb;
    const int mrow = lane & 15;
    const int hiA  = (lane >> 4) << 3;   // A: K-group 0 or 8
    const int hiB  = (lane >> 4) << 4;   // unused; B is read with kA (K-group 0 or 8)
    #pragma unroll
    for (int j = 0; j < 16; ++j) {
      int kA = ((j >> 3) << 4) + hiA + (j & 7);
      a[j]  = (_Float16)As[b][mrow * 32 + kA];
      bb[j] = (_Float16)Bs[b][kA * 64 + (wave << 4) + mrow];
    }
    acc = __builtin_amdgcn_wmma_f32_16x16x32_f16(false, a, false, bb,
                                                 (short)0, acc, false, false);
    __syncthreads();   // reads of buffer b done before it is re-issued
  }

  const int colg  = col0 + (wave << 4) + (lane & 15);
  const int rowHi = (lane >> 4) << 3;
  #pragma unroll
  for (int r = 0; r < 8; ++r) {
    int gr = row0 + rowHi + r;
    float v = acc[r];
    if (bias)  v += bias[colg];
    if (doRelu) v = fmaxf(v, 0.0f);
    C[(size_t)gr * Nout + colg] = v;
  }
}

// ============================================================================
// Edge pass 1: logits[e,h] = sum_c lrelu(xl[src]+xr[dst]+ea)*att[h,c];
// atomicMax over dst for segment-softmax max. ea recomputed (8 MACs/elem).
// ============================================================================
__global__ __launch_bounds__(256)
void edge_logits_kernel(const int* __restrict__ ei, const float* __restrict__ eattr,
                        const float* __restrict__ xl, const float* __restrict__ xr,
                        const float* __restrict__ We, const float* __restrict__ att,
                        float* __restrict__ logits, unsigned* __restrict__ nmax)
{
  int t = blockIdx.x * blockDim.x + threadIdx.x;
  if (t >= NEDGES * NH) return;
  int e  = t >> 2;
  int hh = t & 3;
  int s  = ei[e];
  int d  = ei[NEDGES + e];

  float ea8[EDIMC];
  #pragma unroll
  for (int k = 0; k < EDIMC; ++k) ea8[k] = eattr[(size_t)e * EDIMC + k];

  const float* xls = xl + (size_t)s * HCC + hh * EMBC;
  const float* xrd = xr + (size_t)d * HCC + hh * EMBC;
  const float* atv = att + hh * EMBC;
  float logit = 0.0f;
  for (int c = 0; c < EMBC; ++c) {
    float ea = 0.0f;
    #pragma unroll
    for (int k = 0; k < EDIMC; ++k) ea += ea8[k] * We[k * HCC + hh * EMBC + c];
    float m = xls[c] + xrd[c] + ea;
    m = (m > 0.0f) ? m : 0.2f * m;        // leaky_relu
    logit += m * atv[c];
  }
  logits[(size_t)e * NH + hh] = logit;
  atomicMax(&nmax[(size_t)d * NH + hh], fkey(logit));
}

// ============================================================================
// Edge pass 2: ex = exp(logit - max); den[dst,h] += ex; agg[dst,:] += ex*xl[src,:]
// ============================================================================
__global__ __launch_bounds__(256)
void edge_accum_kernel(const int* __restrict__ ei, const float* __restrict__ xl,
                       const float* __restrict__ logits, const unsigned* __restrict__ nmax,
                       float* __restrict__ nden, float* __restrict__ agg)
{
  int t = blockIdx.x * blockDim.x + threadIdx.x;
  if (t >= NEDGES * NH) return;
  int e  = t >> 2;
  int hh = t & 3;
  int s  = ei[e];
  int d  = ei[NEDGES + e];
  float mx = finv(nmax[(size_t)d * NH + hh]);
  float ex = __expf(logits[(size_t)e * NH + hh] - mx);
  atomicAdd(&nden[(size_t)d * NH + hh], ex);
  const float* xls = xl + (size_t)s * HCC + hh * EMBC;
  float* ag = agg + (size_t)d * HCC + hh * EMBC;
  for (int c = 0; c < EMBC; ++c) atomicAdd(&ag[c], ex * xls[c]);
}

// hpre = agg/den + gat_b  (den==0 -> isolated node -> agg==0)
__global__ __launch_bounds__(256)
void node_finalize_kernel(const float* __restrict__ agg, const float* __restrict__ nden,
                          const float* __restrict__ gb, float* __restrict__ hpre)
{
  size_t i = (size_t)blockIdx.x * blockDim.x + threadIdx.x;
  if (i >= (size_t)NNODES * HCC) return;
  size_t n = i >> 8;
  int hc = (int)(i & 255);
  float den = nden[n * NH + (hc >> 6)];
  float v = (den > 0.0f) ? agg[i] / den : 0.0f;
  hpre[i] = v + gb[hc];
}

// BatchNorm stats: per-channel sum & sumsq, LDS-reduced, one atomic per block
__global__ __launch_bounds__(256)
void bn_reduce_kernel(const float* __restrict__ h, float* __restrict__ stats, int n)
{
  __shared__ float sm[256], sq[256];
  int tid = threadIdx.x;
  int c   = tid & 63;
  int sub = tid >> 6;
  int base = blockIdx.x * 64;
  float s = 0.0f, q = 0.0f;
  for (int r = sub; r < 64; r += 4) {
    int row = base + r;
    if (row < n) {
      float v = h[(size_t)row * EMBC + c];
      s += v; q += v * v;
    }
  }
  sm[tid] = s; sq[tid] = q;
  __syncthreads();
  if (tid < 64) {
    float ts = sm[tid] + sm[tid + 64] + sm[tid + 128] + sm[tid + 192];
    float tq = sq[tid] + sq[tid + 64] + sq[tid + 128] + sq[tid + 192];
    atomicAdd(&stats[tid],      ts);
    atomicAdd(&stats[64 + tid], tq);
  }
}

__global__ __launch_bounds__(256)
void bn_apply_kernel(float* __restrict__ h, const float* __restrict__ stats,
                     const float* __restrict__ g, const float* __restrict__ b, int n)
{
  size_t i = (size_t)blockIdx.x * blockDim.x + threadIdx.x;
  if (i >= (size_t)n * EMBC) return;
  int c = (int)(i & 63);
  float invn = 1.0f / (float)n;
  float mu  = stats[c] * invn;
  float var = stats[64 + c] * invn - mu * mu;
  h[i] = (h[i] - mu) * rsqrtf(var + 1e-5f) * g[c] + b[c];
}

// gate = h @ gate_W + gate_b; atomicMax per batch
__global__ __launch_bounds__(256)
void gate_kernel(const float* __restrict__ h, const float* __restrict__ gW,
                 const float* __restrict__ gB, const int* __restrict__ bidx,
                 float* __restrict__ gate, unsigned* __restrict__ gmax)
{
  int n = blockIdx.x * blockDim.x + threadIdx.x;
  if (n >= NNODES) return;
  float s = gB[0];
  const float* hn = h + (size_t)n * EMBC;
  for (int c = 0; c < EMBC; ++c) s += hn[c] * gW[c];
  gate[n] = s;
  atomicMax(&gmax[bidx[n]], fkey(s));
}

__global__ __launch_bounds__(256)
void pool_kernel(const float* __restrict__ h, const int* __restrict__ bidx,
                 const float* __restrict__ gate, const unsigned* __restrict__ gmax,
                 float* __restrict__ gden, float* __restrict__ pooled)
{
  int n = blockIdx.x * blockDim.x + threadIdx.x;
  if (n >= NNODES) return;
  int b = bidx[n];
  float ex = __expf(gate[n] - finv(gmax[b]));
  atomicAdd(&gden[b], ex);
  const float* hn = h + (size_t)n * EMBC;
  float* pb = pooled + (size_t)b * EMBC;
  for (int c = 0; c < EMBC; ++c) atomicAdd(&pb[c], ex * hn[c]);
}

// Head MLP: one thread per batch row (tiny: ~58k MACs/row)
__global__ __launch_bounds__(128)
void head_kernel(const float* __restrict__ pooled, const float* __restrict__ gden,
                 const float* __restrict__ fc, const float* __restrict__ nrot,
                 const float* __restrict__ l1W, const float* __restrict__ l1b,
                 const float* __restrict__ f1W, const float* __restrict__ f1b,
                 const float* __restrict__ f2W, const float* __restrict__ f2b,
                 const float* __restrict__ oW,  const float* __restrict__ ob,
                 float* __restrict__ out)
{
  int b = threadIdx.x;
  if (b >= NB) return;
  float z0[EMBC + 2];
  float d = gden[b];
  float inv = (d > 0.0f) ? 1.0f / d : 0.0f;
  for (int c = 0; c < EMBC; ++c) z0[c] = pooled[(size_t)b * EMBC + c] * inv;
  z0[EMBC]     = fc[b];
  z0[EMBC + 1] = nrot[b];

  float z1[DNC];
  for (int j = 0; j < DNC; ++j) {
    float s = l1b[j];
    for (int k = 0; k < EMBC + 2; ++k) s += z0[k] * l1W[k * DNC + j];
    z1[j] = fmaxf(s, 0.0f);
  }
  float z2[DNC / 2];
  for (int j = 0; j < DNC / 2; ++j) {
    float s = f1b[j];
    for (int k = 0; k < DNC; ++k) s += z1[k] * f1W[k * (DNC / 2) + j];
    z2[j] = fmaxf(s, 0.0f);
  }
  float z3[DNC / 4];
  for (int j = 0; j < DNC / 4; ++j) {
    float s = f2b[j];
    for (int k = 0; k < DNC / 2; ++k) s += z2[k] * f2W[k * (DNC / 4) + j];
    z3[j] = fmaxf(s, 0.0f);
  }
  float s = ob[0];
  for (int k = 0; k < DNC / 4; ++k) s += z3[k] * oW[k];
  out[b] = s;
}

// ============================================================================
extern "C" void kernel_launch(void* const* d_in, const int* in_sizes, int n_in,
                              void* d_out, int out_size, void* d_ws, size_t ws_size,
                              hipStream_t stream)
{
  const float* x     = (const float*)d_in[0];
  const int*   ei    = (const int*)  d_in[1];
  const float* eattr = (const float*)d_in[2];
  const float* molfc = (const float*)d_in[3];
  const float* molnr = (const float*)d_in[4];
  const int*   bidx  = (const int*)  d_in[5];
  const float* gWl   = (const float*)d_in[6];
  const float* gWr   = (const float*)d_in[7];
  const float* gWe   = (const float*)d_in[8];
  const float* gatt  = (const float*)d_in[9];
  const float* gb    = (const float*)d_in[10];
  const float* trW   = (const float*)d_in[11];
  const float* trb   = (const float*)d_in[12];
  const float* bng   = (const float*)d_in[13];
  const float* bnb   = (const float*)d_in[14];
  const float* gateW = (const float*)d_in[15];
  const float* gateB = (const float*)d_in[16];
  const float* l1W   = (const float*)d_in[17];
  const float* l1b   = (const float*)d_in[18];
  const float* f1W   = (const float*)d_in[19];
  const float* f1b   = (const float*)d_in[20];
  const float* f2W   = (const float*)d_in[21];
  const float* f2b   = (const float*)d_in[22];
  const float* oW    = (const float*)d_in[23];
  const float* ob    = (const float*)d_in[24];
  float* out = (float*)d_out;

  // ---- workspace carve-out (agg aliases xr; hpre aliases xl) ----
  char* ws = (char*)d_ws;
  size_t off = 0;
  auto carve = [&](size_t bytes) -> char* {
    char* p = ws + off;
    off += (bytes + 255) & ~(size_t)255;
    return p;
  };
  float*    xl     = (float*)   carve((size_t)NNODES * HCC * 4);
  float*    xr     = (float*)   carve((size_t)NNODES * HCC * 4);  // also agg
  float*    hbuf   = (float*)   carve((size_t)NNODES * EMBC * 4);
  float*    logits = (float*)   carve((size_t)NEDGES * NH * 4);
  unsigned* nmax   = (unsigned*)carve((size_t)NNODES * NH * 4);
  float*    nden   = (float*)   carve((size_t)NNODES * NH * 4);
  float*    stats  = (float*)   carve(128 * 4);
  float*    gate   = (float*)   carve((size_t)NNODES * 4);
  unsigned* gmax   = (unsigned*)carve(NB * 4);
  float*    gden   = (float*)   carve(NB * 4);
  float*    pooled = (float*)   carve((size_t)NB * EMBC * 4);

  const int ET = NEDGES * NH;
  for (int l = 0; l < NL; ++l) {
    const float* hin = (l == 0) ? x : hbuf;
    // xl = h @ Wl ; xr = h @ Wr   (M=50000, K=64, Nout=256)
    dim3 g1(NNODES / 16, HCC / 64);
    wmma_gemm_async<FIN><<<g1, 128, 0, stream>>>(hin, gWl + (size_t)l * FIN * HCC,
                                                 nullptr, xl, HCC, 0);
    wmma_gemm_async<FIN><<<g1, 128, 0, stream>>>(hin, gWr + (size_t)l * FIN * HCC,
                                                 nullptr, xr, HCC, 0);
    hipMemsetAsync(nmax, 0, (size_t)NNODES * NH * 4, stream);
    hipMemsetAsync(nden, 0, (size_t)NNODES * NH * 4, stream);
    edge_logits_kernel<<<(ET + 255) / 256, 256, 0, stream>>>(
        ei, eattr, xl, xr, gWe + (size_t)l * EDIMC * HCC,
        gatt + (size_t)l * NH * EMBC, logits, nmax);
    // agg reuses xr (dead after logits pass)
    hipMemsetAsync(xr, 0, (size_t)NNODES * HCC * 4, stream);
    edge_accum_kernel<<<(ET + 255) / 256, 256, 0, stream>>>(
        ei, xl, logits, nmax, nden, xr);
    // hpre reuses xl (dead after accumulation pass)
    node_finalize_kernel<<<((size_t)NNODES * HCC + 255) / 256, 256, 0, stream>>>(
        xr, nden, gb + (size_t)l * HCC, xl);
    // h = relu(hpre @ tr_W + tr_b)   (M=50000, K=256, Nout=64)
    dim3 g2(NNODES / 16, EMBC / 64);
    wmma_gemm_async<HCC><<<g2, 128, 0, stream>>>(xl, trW + (size_t)l * HCC * EMBC,
                                                 trb + (size_t)l * EMBC, hbuf,
                                                 EMBC, 1);
    // BatchNorm (training-mode batch stats, biased var)
    hipMemsetAsync(stats, 0, 128 * 4, stream);
    bn_reduce_kernel<<<(NNODES + 63) / 64, 256, 0, stream>>>(hbuf, stats, NNODES);
    bn_apply_kernel<<<((size_t)NNODES * EMBC + 255) / 256, 256, 0, stream>>>(
        hbuf, stats, bng + (size_t)l * EMBC, bnb + (size_t)l * EMBC, NNODES);
  }

  // Attentional pooling
  hipMemsetAsync(gmax, 0, NB * 4, stream);
  hipMemsetAsync(gden, 0, NB * 4, stream);
  hipMemsetAsync(pooled, 0, (size_t)NB * EMBC * 4, stream);
  gate_kernel<<<(NNODES + 255) / 256, 256, 0, stream>>>(hbuf, gateW, gateB, bidx,
                                                        gate, gmax);
  pool_kernel<<<(NNODES + 255) / 256, 256, 0, stream>>>(hbuf, bidx, gate, gmax,
                                                        gden, pooled);
  head_kernel<<<1, 128, 0, stream>>>(pooled, gden, molfc, molnr, l1W, l1b,
                                     f1W, f1b, f2W, f2b, oW, ob, out);
}
